// BlockDiffusionDecoder_23450521436795
// MI455X (gfx1250) — matrix operand order repaired and checked by reading, not compiled, not run on hardware
//
#include <hip/hip_runtime.h>

// ---------------------------------------------------------------------------
// BD3LM block-diffusion attention layer for MI455X (gfx1250, wave32, WMMA).
//  pass 0: x -> bf16 once; W* -> bf16 transposed once (memory-bound)
//  pass 1: 3x GEMM (async-LDS double-buffered, bf16 WMMA) -> Qh,Kh,VhT
//  pass 2: masked flash attention, S^T=K@Q^T (in-lane softmax), O+=P@V,
//          with mask-free interior tiles + masked boundary tiles
//  pass 3: GEMM ctx@Wo + bo -> f32 out
// Workspace: Qh,Kh,VhT,ctx (4x16MB) + xb(16MB) + 4x W^T(2MB) = 88MB.
// ---------------------------------------------------------------------------

typedef __attribute__((ext_vector_type(16))) __bf16         v16bf;
typedef __attribute__((ext_vector_type(8)))  float          v8f;
typedef __attribute__((ext_vector_type(4)))  float          f4;
typedef __attribute__((ext_vector_type(8)))  unsigned short us8;
typedef __attribute__((ext_vector_type(4)))  unsigned short us4;
typedef __attribute__((ext_vector_type(16))) unsigned short us16;

struct us8x2 { us8 lo, hi; };

__device__ inline unsigned short f2bf(float f) {
  unsigned int u = __builtin_bit_cast(unsigned int, f);
  u += 0x7FFFu + ((u >> 16) & 1u);  // round-to-nearest-even
  return (unsigned short)(u >> 16);
}
__device__ inline v16bf mk16(us8 a, us8 b) {
  us8x2 t{a, b};
  return __builtin_bit_cast(v16bf, t);
}
__device__ inline v8f vzero() {
  v8f z = {0.f, 0.f, 0.f, 0.f, 0.f, 0.f, 0.f, 0.f};
  return z;
}
__device__ inline v8f wmma_bf16(v16bf a, v16bf b, v8f c) {
  return __builtin_amdgcn_wmma_f32_16x16x32_bf16(false, a, false, b,
                                                 (short)0, c, false, false);
}
// swap lanes L <-> L+16 (group-of-32 swizzle, xor_mask=0x10)
__device__ inline float xor16f(float v) {
  return __builtin_bit_cast(
      float, __builtin_amdgcn_ds_swizzle(__builtin_bit_cast(int, v), 0x401F));
}
__device__ inline float bcastf(float v, int srcLane) {
  return __builtin_bit_cast(
      float, __builtin_amdgcn_ds_bpermute(srcLane << 2,
                                          __builtin_bit_cast(int, v)));
}
// async DMA: 16B global -> LDS, tracked by ASYNCcnt (no VGPR data path)
__device__ inline void async_cp16(unsigned lds_off, unsigned long long gaddr) {
  asm volatile("global_load_async_to_lds_b128 %0, %1, off"
               :: "v"(lds_off), "v"(gaddr) : "memory");
}
__device__ inline void async_wait0() {
  asm volatile("s_wait_asynccnt 0x0" ::: "memory");
}

// ---------------------------------------------------------------------------
// pre-pass: x (f32) -> bf16, flat
__global__ __launch_bounds__(256) void cvt_bf16_kernel(
    const float* __restrict__ X, unsigned short* __restrict__ Xb) {
  size_t i = ((size_t)blockIdx.x * 256 + threadIdx.x) * 4;
  f4 v = *(const f4*)(X + i);
  us4 h;
#pragma unroll
  for (int j = 0; j < 4; ++j) h[j] = f2bf(v[j]);
  *(us4*)(Xb + i) = h;
}

// pre-pass: W[k][n] f32 -> WT[n][k] bf16 (32x32 LDS tile transpose)
__global__ __launch_bounds__(256) void wt_bf16_kernel(
    const float* __restrict__ W, unsigned short* __restrict__ WT) {
  __shared__ float t[32][33];
  const int n0 = blockIdx.x * 32, k0 = blockIdx.y * 32;
  const int tx = threadIdx.x & 31, ty = threadIdx.x >> 5;
#pragma unroll
  for (int i = 0; i < 4; ++i)
    t[ty + i * 8][tx] = W[(size_t)(k0 + ty + i * 8) * 1024 + n0 + tx];
  __syncthreads();
#pragma unroll
  for (int i = 0; i < 4; ++i)
    WT[(size_t)(n0 + ty + i * 8) * 1024 + k0 + tx] = f2bf(t[tx][ty + i * 8]);
}

// ---------------------------------------------------------------------------
// GEMM: Y = Xb[8192x1024](bf16) @ WT^T + bias; WT is [n][k] bf16.
// 256 thr, 128x128 block tile, k-step 64, double-buffered async-LDS staging.
// MODE 0: bf16 per-head [b][h][t][64], (acc+bias)*scale   (Q,K)
// MODE 1: bf16 per-head transposed [b][h][64][t]          (V^T)
// MODE 2: f32 [m][n]                                      (final out)
// ---------------------------------------------------------------------------
#define LDS_STRIDE 72           // 64 halfs + 8 pad (36 dwords: conflict-free)
#define BUF_HALFS  (128 * LDS_STRIDE)

template <int MODE>
__global__ __launch_bounds__(256) void gemm_tile_kernel(
    const unsigned short* __restrict__ Xb, const unsigned short* __restrict__ WT,
    const float* __restrict__ bias, void* __restrict__ outp, float scale) {
  __shared__ __align__(16) unsigned short Xs[2 * BUF_HALFS];
  __shared__ __align__(16) unsigned short Ws[2 * BUF_HALFS];
  const int tid  = threadIdx.x;
  const int lane = tid & 31, wave = tid >> 5;
  const int m0 = blockIdx.x * 128, n0 = blockIdx.y * 128;
  const int mb = (wave >> 2) * 64, nb = (wave & 3) * 32;  // 2x4 wave grid

  // per-thread staging descriptors: 4 chunks of 16B each for X and W tiles
  unsigned long long xg[4], wg[4];
  unsigned xl[4], wl[4];
#pragma unroll
  for (int it = 0; it < 4; ++it) {
    int idx = tid + it * 256;
    int r = idx >> 3, c = idx & 7;  // 128 rows x 8 chunks (64 halfs)
    xg[it] = (unsigned long long)(const void*)(Xb + (size_t)(m0 + r) * 1024 + c * 8);
    wg[it] = (unsigned long long)(const void*)(WT + (size_t)(n0 + r) * 1024 + c * 8);
    unsigned l = (unsigned)(r * LDS_STRIDE + c * 8) * 2u;
    xl[it] = (unsigned)(unsigned long long)(const void*)Xs + l;
    wl[it] = (unsigned)(unsigned long long)(const void*)Ws + l;
  }

  v8f acc[4][2];
#pragma unroll
  for (int i = 0; i < 4; ++i)
#pragma unroll
    for (int j = 0; j < 2; ++j) acc[i][j] = vzero();

  const int kb = (lane >> 4) << 3;  // A-frag K sub-offset {0,8}
  const int ko = (lane >> 4) << 3;  // B-frag K sub-offset {0,8}

  // prologue: stage k-chunk 0 into buffer 0
#pragma unroll
  for (int it = 0; it < 4; ++it) async_cp16(xl[it], xg[it]);
#pragma unroll
  for (int it = 0; it < 4; ++it) async_cp16(wl[it], wg[it]);

  for (int i = 0; i < 16; ++i) {           // 16 x (k-step 64)
    async_wait0();
    __syncthreads();                       // buffer i&1 fully staged
    if (i + 1 < 16) {                      // stage next into other buffer
      unsigned bo = (unsigned)(((i + 1) & 1) * BUF_HALFS * 2);
      unsigned long long go = (unsigned long long)(i + 1) * 128;  // 64 halfs
#pragma unroll
      for (int it = 0; it < 4; ++it) async_cp16(xl[it] + bo, xg[it] + go);
#pragma unroll
      for (int it = 0; it < 4; ++it) async_cp16(wl[it] + bo, wg[it] + go);
    }
    const unsigned short* Xc = Xs + (i & 1) * BUF_HALFS;
    const unsigned short* Wc = Ws + (i & 1) * BUF_HALFS;
#pragma unroll
    for (int kc = 0; kc < 2; ++kc) {
      v16bf af[4], bfr[2];
#pragma unroll
      for (int ms = 0; ms < 4; ++ms) {
        const unsigned short* p =
            Xc + (mb + ms * 16 + (lane & 15)) * LDS_STRIDE + kc * 32 + kb;
        af[ms] = mk16(*(const us8*)p, *(const us8*)(p + 16));
      }
#pragma unroll
      for (int ns = 0; ns < 2; ++ns) {
        const unsigned short* p =
            Wc + (nb + ns * 16 + (lane & 15)) * LDS_STRIDE + kc * 32 + ko;
        bfr[ns] = mk16(*(const us8*)p, *(const us8*)(p + 16));
      }
#pragma unroll
      for (int ms = 0; ms < 4; ++ms)
#pragma unroll
        for (int ns = 0; ns < 2; ++ns)
          acc[ms][ns] = wmma_bf16(af[ms], bfr[ns], acc[ms][ns]);
    }
    __syncthreads();                       // done reading buffer i&1
  }

  // ---- epilogue ----
#pragma unroll
  for (int ns = 0; ns < 2; ++ns) {
    const int ng = blockIdx.y * 128 + nb + ns * 16 + (lane & 15);
    const float bv = bias[ng];
    const int h = ng >> 6, d = ng & 63;
#pragma unroll
    for (int ms = 0; ms < 4; ++ms) {
      const int mg0 = blockIdx.x * 128 + mb + ms * 16 + ((lane >> 4) << 3);
      v8f a = acc[ms][ns];
      if (MODE == 2) {
        float* o = (float*)outp;
#pragma unroll
        for (int j = 0; j < 8; ++j) o[(size_t)(mg0 + j) * 1024 + ng] = a[j] + bv;
      } else if (MODE == 0) {
        unsigned short* o = (unsigned short*)outp;
#pragma unroll
        for (int j = 0; j < 8; ++j) {
          int mg = mg0 + j, bb = mg >> 11, t = mg & 2047;
          o[(((size_t)(bb * 16 + h) * 2048 + t) * 64 + d)] =
              f2bf((a[j] + bv) * scale);
        }
      } else {  // MODE 1: V^T, 8 consecutive t per lane -> one b128 store
        unsigned short* o = (unsigned short*)outp;
        int bb = mg0 >> 11, t0 = mg0 & 2047;
        us8 pk;
#pragma unroll
        for (int j = 0; j < 8; ++j) pk[j] = f2bf((a[j] + bv) * scale);
        *(us8*)(o + ((size_t)((bb * 16 + h) * 64 + d) * 2048 + t0)) = pk;
      }
    }
  }
}

// ---------------------------------------------------------------------------
// Masked flash attention. 4 waves/block; wave w owns 16 q rows.
// S^T = K @ Q^T (softmax stats in-lane), O += P @ V via the P^T->A identity.
// Interior kv tiles are provably mask-free for all lanes; only the diagonal
// and the last boundary tile(s) pay the mask VALU cost.
// ---------------------------------------------------------------------------
__global__ __launch_bounds__(128) void bd3lm_attn_kernel(
    const unsigned short* __restrict__ Qh, const unsigned short* __restrict__ Kh,
    const unsigned short* __restrict__ VhT, unsigned short* __restrict__ ctx) {
  const int lane = threadIdx.x & 31, wave = threadIdx.x >> 5;
  const int q0 = blockIdx.x * 64 + wave * 16;
  const int h = blockIdx.y, b = blockIdx.z;
  const int lj0 = (lane >> 4) << 3;  // C-layout row base {0,8}

  const unsigned short* Qp = Qh + ((size_t)(b * 16 + h) * 2048 + q0) * 64;
  const unsigned short* Kb = Kh + (size_t)(b * 16 + h) * 2048 * 64;
  const unsigned short* Vb = VhT + (size_t)(b * 16 + h) * 64 * 2048;

  // Q^T B-frags (hd chunks 0..31, 32..63), reused across all kv tiles
  v16bf qf[2];
#pragma unroll
  for (int c = 0; c < 2; ++c) {
    const unsigned short* p = Qp + (lane & 15) * 64 + c * 32 + ((lane >> 4) << 3);
    qf[c] = mk16(*(const us8*)p, *(const us8*)(p + 16));
  }

  v8f o[4];
#pragma unroll
  for (int c = 0; c < 4; ++c) o[c] = vzero();
  float mrow = -1.0e30f, lrow = 0.0f;

  // BD3LM mask as two per-lane kv intervals [I0,I0+L0) U [I1,I1+L1)
  const int qv = q0 + (lane & 15);
  int I0, I1;
  unsigned L0, L1;
  if (qv < 1024) {  // xt row: own diagonal block + strictly-earlier x0 blocks
    I0 = qv & ~3; L0 = 4u;
    I1 = 1024;    L1 = (unsigned)(qv & ~3);
  } else {          // x0 row: x0 blocks up to and including own
    I0 = 1024;    L0 = (unsigned)(((qv - 1024) & ~3) + 4);
    I1 = 0;       L1 = 0u;
  }

  auto tile = [&](int kvt, const bool masked) {
    // K A-frags: [kv-sub s][hd chunk c]
    v16bf ak[2][2];
#pragma unroll
    for (int s = 0; s < 2; ++s)
#pragma unroll
      for (int c = 0; c < 2; ++c) {
        const unsigned short* p = Kb + (size_t)(kvt + s * 16 + (lane & 15)) * 64 +
                                  c * 32 + lj0;
        ak[s][c] = mk16(*(const us8*)p, *(const us8*)(p + 16));
      }
    // S^T tiles (kv x q), f32
    v8f st[2];
#pragma unroll
    for (int s = 0; s < 2; ++s) {
      v8f cacc = vzero();
      cacc = wmma_bf16(ak[s][0], qf[0], cacc);
      cacc = wmma_bf16(ak[s][1], qf[1], cacc);
      st[s] = cacc;
    }
    // mask (boundary tiles only): two unsigned range checks per element
    if (masked) {
#pragma unroll
      for (int s = 0; s < 2; ++s)
#pragma unroll
        for (int j = 0; j < 8; ++j) {
          int kv = kvt + s * 16 + lj0 + j;
          bool ok = ((unsigned)(kv - I0) < L0) || ((unsigned)(kv - I1) < L1);
          st[s][j] = ok ? st[s][j] : -3.0e38f;
        }
    }
    // online softmax: stats per q column (in-lane + one xor16)
    float tmax = -3.0e38f;
#pragma unroll
    for (int s = 0; s < 2; ++s)
#pragma unroll
      for (int j = 0; j < 8; ++j) tmax = fmaxf(tmax, st[s][j]);
    tmax = fmaxf(tmax, xor16f(tmax));
    float mnew = fmaxf(mrow, tmax);
    float rs = 0.0f;
#pragma unroll
    for (int s = 0; s < 2; ++s)
#pragma unroll
      for (int j = 0; j < 8; ++j) {
        float p = __expf(st[s][j] - mnew);
        st[s][j] = p;
        rs += p;
      }
    rs += xor16f(rs);
    float fac = __expf(mrow - mnew);
    lrow = lrow * fac + rs;
    mrow = mnew;
    // rescale O (factor per C-layout row, via lane broadcast)
#pragma unroll
    for (int j = 0; j < 8; ++j) {
      float fj = bcastf(fac, lj0 + j);
#pragma unroll
      for (int c = 0; c < 4; ++c) o[c][j] *= fj;
    }
    // P^T C-tiles -> A operand of P (16q x 32kv): bf16 convert, no lane moves
    us16 pu;
#pragma unroll
    for (int j = 0; j < 8; ++j) {
      pu[j] = f2bf(st[0][j]);
      pu[j + 8] = f2bf(st[1][j]);
    }
    v16bf pa = __builtin_bit_cast(v16bf, pu);
    // O += P @ V  (B = V from V^T, kv in table order per lane)
#pragma unroll
    for (int c = 0; c < 4; ++c) {
      const unsigned short* p =
          Vb + (size_t)(c * 16 + (lane & 15)) * 2048 + kvt + ((lane >> 4) << 3);
      v16bf vb = mk16(*(const us8*)p, *(const us8*)(p + 16));
      o[c] = wmma_bf16(pa, vb, o[c]);
    }
  };

  // kv tile schedule: mask-free interior, masked boundary.
  // min valid-prefix over the wave's 16 q rows:
  //   xt rows: kv < 1024 + (qv&~3)        -> min 1024 + q0
  //   x0 rows: kv < 1024 + ((qv-1024)&~3)+4 -> min q0 + 4
  if (q0 < 1024) {
    tile(q0 & ~31, true);  // xt diagonal block tile (always partially masked)
    int e1 = 1024 + (((q0 + 16) + 31) & ~31);
    int kvt = 1024;
    for (; kvt + 32 <= 1024 + q0; kvt += 32) tile(kvt, false);
    for (; kvt < e1; kvt += 32) tile(kvt, true);
  } else {
    int e1 = 1024 + (((q0 - 1024 + 16) + 31) & ~31);
    int kvt = 1024;
    for (; kvt + 32 <= q0 + 4; kvt += 32) tile(kvt, false);
    for (; kvt < e1; kvt += 32) tile(kvt, true);
  }

  // finalize: O /= l, store ctx bf16 [b][t][h*64+d]
  float linv = 1.0f / lrow;
#pragma unroll
  for (int j = 0; j < 8; ++j) {
    float lj = bcastf(linv, lj0 + j);
    int t = q0 + lj0 + j;
#pragma unroll
    for (int c = 0; c < 4; ++c)
      ctx[(size_t)(b * 2048 + t) * 1024 + h * 64 + c * 16 + (lane & 15)] =
          f2bf(o[c][j] * lj);
  }
}

// ---------------------------------------------------------------------------
extern "C" void kernel_launch(void* const* d_in, const int* in_sizes, int n_in,
                              void* d_out, int out_size, void* d_ws,
                              size_t ws_size, hipStream_t stream) {
  (void)in_sizes; (void)n_in; (void)out_size; (void)ws_size;
  const float* x  = (const float*)d_in[0];
  const float* Wq = (const float*)d_in[1];
  const float* bq = (const float*)d_in[2];
  const float* Wk = (const float*)d_in[3];
  const float* bk = (const float*)d_in[4];
  const float* Wv = (const float*)d_in[5];
  const float* bv = (const float*)d_in[6];
  const float* Wo = (const float*)d_in[7];
  const float* bo = (const float*)d_in[8];
  // block_size (d_in[9]) fixed at 4, baked into the mask predicate.

  const size_t E = 8192ull * 1024ull;   // activation elements
  const size_t WE = 1024ull * 1024ull;  // weight elements
  unsigned short* Qh  = (unsigned short*)d_ws;
  unsigned short* Kh  = Qh + E;
  unsigned short* VhT = Kh + E;
  unsigned short* ctx = VhT + E;
  unsigned short* xb  = ctx + E;
  unsigned short* WTq = xb + E;
  unsigned short* WTk = WTq + WE;
  unsigned short* WTv = WTk + WE;
  unsigned short* WTo = WTv + WE;

  cvt_bf16_kernel<<<8192, 256, 0, stream>>>(x, xb);
  dim3 gt(32, 32);
  wt_bf16_kernel<<<gt, 256, 0, stream>>>(Wq, WTq);
  wt_bf16_kernel<<<gt, 256, 0, stream>>>(Wk, WTk);
  wt_bf16_kernel<<<gt, 256, 0, stream>>>(Wv, WTv);
  wt_bf16_kernel<<<gt, 256, 0, stream>>>(Wo, WTo);

  dim3 gg(64, 8);
  gemm_tile_kernel<0><<<gg, 256, 0, stream>>>(xb, WTq, bq, Qh, 0.125f);
  gemm_tile_kernel<0><<<gg, 256, 0, stream>>>(xb, WTk, bk, Kh, 1.0f);
  gemm_tile_kernel<1><<<gg, 256, 0, stream>>>(xb, WTv, bv, VhT, 1.0f);
  bd3lm_attn_kernel<<<dim3(32, 16, 4), 128, 0, stream>>>(Qh, Kh, VhT, ctx);
  gemm_tile_kernel<2><<<gg, 256, 0, stream>>>(ctx, WTo, bo, d_out, 1.0f);
}
